// DN4Fast_10668698763885
// MI455X (gfx1250) — matrix operand order repaired and checked by reading, not compiled, not run on hardware
//
#include <hip/hip_runtime.h>
#include <hip/hip_bf16.h>

typedef _Float16 f16;
typedef __attribute__((ext_vector_type(16))) _Float16 v16h;
typedef __attribute__((ext_vector_type(8)))  _Float16 v8h;
typedef __attribute__((ext_vector_type(8)))  float    v8f;

__device__ __forceinline__ v8f wmma16x16x32(v16h a, v16h b, v8f c) {
  // (neg_a, A, neg_b, B, c_mod, C, reuse_a, reuse_b)
  return __builtin_amdgcn_wmma_f32_16x16x32_f16(false, a, false, b, (short)0, c,
                                                false, false);
}

// ----------------------------- utilities -----------------------------------
__global__ void zero_f32_k(float* p, long n) {
  long i = (long)blockIdx.x * blockDim.x + threadIdx.x;
  if (i < n) p[i] = 0.f;
}
__global__ void zero_f16_k(f16* p, long n) {
  long i = (long)blockIdx.x * blockDim.x + threadIdx.x;
  if (i < n) p[i] = (f16)0.f;
}

// NCHW f32 -> NHWC f16
__global__ void to_nhwc_f16_k(const float* __restrict__ in, f16* __restrict__ out,
                              int N, int C, int H, int W) {
  long total = (long)N * C * H * W;
  long i = (long)blockIdx.x * blockDim.x + threadIdx.x;
  if (i >= total) return;
  int c = (int)(i % C); long t = i / C;
  int x = (int)(t % W); t /= W;
  int y = (int)(t % H); int n = (int)(t / H);
  out[i] = (f16)in[(((long)n * C + c) * H + y) * W + x];
}

// Pack OIHW f32 weights into WMMA A-fragment lane layout, f16.
// Packed index: ((ot*NKC + kc)*32 + lane)*16 + h
// A 16x32 layout (contiguous K order chosen here, matched by the B loads): lane<16 holds M=lane, K b..b+15, b=0;
// lane>=16 same M, b=16.  K ordering: K = kpos*CIN + cin  (kpos = ky*3+kx).
__global__ void pack_w_k(const float* __restrict__ w, f16* __restrict__ wp, int CIN) {
  int NKC = (CIN * 9 + 31) / 32;
  int total = 4 * NKC * 32 * 16;
  int i = blockIdx.x * blockDim.x + threadIdx.x;
  if (i >= total) return;
  int h = i & 15;
  int lane = (i >> 4) & 31;
  int t = i >> 9;
  int kc = t % NKC;
  int ot = t / NKC;
  int o = ot * 16 + (lane & 15);
  int Kl = h + ((lane & 16) ? 16 : 0);
  int K = kc * 32 + Kl;
  float v = 0.f;
  if (K < CIN * 9) {
    int kpos = K / CIN, cin = K % CIN;
    v = w[((long)o * CIN + cin) * 9 + kpos];
  }
  wp[i] = (f16)v;
}

// ----------------------- implicit-GEMM conv (WMMA) --------------------------
// act: NHWC f16, out: NHWC f32.  One wave: 16 positions x 64 output channels.
template <int CIN>
__global__ void conv_wmma_k(const f16* __restrict__ act, const f16* __restrict__ wp,
                            float* __restrict__ out, int N, int H, int W) {
  const int NKC = (CIN * 9 + 31) / 32;     // 1 for CIN=3, 18 for CIN=64
  int lane = threadIdx.x & 31;
  long wave = ((long)blockIdx.x * blockDim.x + threadIdx.x) >> 5;
  long total = (long)N * H * W;
  long p0 = wave * 16;
  if (p0 >= total) return;                 // wave-uniform: EXEC stays full
  int col = lane & 15;
  int hsel = lane >> 4;                    // which K-half of the chunk
  long p = p0 + col;
  bool pval = p < total;
  if (!pval) p = total - 1;                // clamp, keep loads in-bounds
  int n = (int)(p / ((long)H * W));
  int rem = (int)(p % ((long)H * W));
  int y = rem / W, x = rem % W;

  v8f acc0 = {}, acc1 = {}, acc2 = {}, acc3 = {};
  for (int kc = 0; kc < NKC; ++kc) {
    v16h bf = {};
    if (CIN == 64) {
      int K = kc * 32 + hsel * 16;         // 16 contiguous K = 16 contiguous cin
      int kpos = K >> 6;
      int cin0 = K & 63;
      int iy = y + kpos / 3 - 1, ix = x + kpos % 3 - 1;
      if (iy >= 0 && iy < H && ix >= 0 && ix < W)
        bf = *(const v16h*)(act + (((long)n * H + iy) * W + ix) * 64 + cin0);
    } else {
      for (int h = 0; h < 16; ++h) {
        int K = kc * 32 + hsel * 16 + h;
        if (K < CIN * 9) {
          int kpos = K / CIN, cin = K % CIN;
          int iy = y + kpos / 3 - 1, ix = x + kpos % 3 - 1;
          if (iy >= 0 && iy < H && ix >= 0 && ix < W)
            bf[h] = act[(((long)n * H + iy) * W + ix) * CIN + cin];
        }
      }
    }
    const f16* wbase = wp + (long)kc * 512 + lane * 16;
    v16h a0 = *(const v16h*)(wbase);
    v16h a1 = *(const v16h*)(wbase + (long)NKC * 512);
    v16h a2 = *(const v16h*)(wbase + (long)NKC * 1024);
    v16h a3 = *(const v16h*)(wbase + (long)NKC * 1536);
    acc0 = wmma16x16x32(a0, bf, acc0);
    acc1 = wmma16x16x32(a1, bf, acc1);
    acc2 = wmma16x16x32(a2, bf, acc2);
    acc3 = wmma16x16x32(a3, bf, acc3);
  }
  if (pval) {
    // D layout: lane holds N=col, rows r + 8*hsel -> channels contiguous per 8
    float* dst = out + (((long)n * H + y) * W + x) * 64 + hsel * 8;
    *(v8f*)(dst)      = acc0;
    *(v8f*)(dst + 16) = acc1;
    *(v8f*)(dst + 32) = acc2;
    *(v8f*)(dst + 48) = acc3;
  }
}

// ------------------------------ batch norm ----------------------------------
__global__ void bn_stats_k(const float* __restrict__ x, float* __restrict__ stats,
                           long rows) {
  __shared__ float ssum[256], ssq[256];
  int c = threadIdx.x & 63;
  int sub = threadIdx.x >> 6;
  float sum = 0.f, sq = 0.f;
  for (long r = (long)blockIdx.x * 4 + sub; r < rows; r += (long)gridDim.x * 4) {
    float v = x[r * 64 + c];
    sum += v; sq += v * v;
  }
  ssum[threadIdx.x] = sum; ssq[threadIdx.x] = sq;
  __syncthreads();
  if (sub == 0) {
    for (int k = 1; k < 4; ++k) { sum += ssum[k * 64 + c]; sq += ssq[k * 64 + c]; }
    atomicAdd(&stats[c], sum);
    atomicAdd(&stats[64 + c], sq);
  }
}

template <bool POOL>
__global__ void bn_act_k(const float* __restrict__ x, const float* __restrict__ stats,
                         const float* __restrict__ g, const float* __restrict__ bb,
                         f16* __restrict__ out, int N, int H, int W, float inv_rows) {
  int Ho = POOL ? H / 2 : H, Wo = POOL ? W / 2 : W;
  long total = (long)N * Ho * Wo * 64;
  long i = (long)blockIdx.x * blockDim.x + threadIdx.x;
  if (i >= total) return;
  int c = (int)(i & 63); long t = i >> 6;
  int xo = (int)(t % Wo); t /= Wo;
  int yo = (int)(t % Ho); int n = (int)(t / Ho);
  float mean = stats[c] * inv_rows;
  float var = stats[64 + c] * inv_rows - mean * mean;   // biased variance
  float sc = g[c] * rsqrtf(var + 1e-5f);
  float sh = bb[c] - mean * sc;
  float r;
  if (POOL) {
    const float* base = x + (((long)n * H + yo * 2) * W + xo * 2) * 64 + c;
    float v00 = base[0], v01 = base[64];
    float v10 = base[(long)W * 64], v11 = base[(long)W * 64 + 64];
    float a = sc * v00 + sh; a = a > 0.f ? a : 0.2f * a;
    float b2 = sc * v01 + sh; b2 = b2 > 0.f ? b2 : 0.2f * b2;
    float c2 = sc * v10 + sh; c2 = c2 > 0.f ? c2 : 0.2f * c2;
    float d2 = sc * v11 + sh; d2 = d2 > 0.f ? d2 : 0.2f * d2;
    r = fmaxf(fmaxf(a, b2), fmaxf(c2, d2));
  } else {
    float v = x[(((long)n * H + yo) * W + xo) * 64 + c];
    r = sc * v + sh; r = r > 0.f ? r : 0.2f * r;
  }
  out[i] = (f16)r;
}

// ------------------- L2-normalized local descriptors ------------------------
__global__ void make_desc_k(const f16* __restrict__ feat, f16* __restrict__ dst_q,
                            f16* __restrict__ dst_s, int N, int HW, int isSupport,
                            int Way, int Shot, int Mpad) {
  long rows = (long)N * HW;
  long row = (long)blockIdx.x * blockDim.x + threadIdx.x;
  if (row >= rows) return;
  const f16* src = feat + row * 64;
  float ss = 0.f;
  for (int c = 0; c < 64; ++c) { float v = (float)src[c]; ss += v * v; }
  float inv = 1.f / fmaxf(sqrtf(ss), 1e-12f);
  f16* dst;
  if (!isSupport) {
    dst = dst_q + row * 64;                 // (b*NQ+q, l, c) layout
  } else {
    int n = (int)(row / HW), l = (int)(row % HW);
    int b = n / (Way * Shot), wsr = n % (Way * Shot);
    int w = wsr / Shot, s = wsr % Shot;
    dst = dst_s + (((long)(b * Way + w) * Mpad) + s * HW + l) * 64;
  }
  for (int c = 0; c < 64; ++c) dst[c] = (f16)((float)src[c] * inv);
}

// ---------------- fused similarity GEMM + top-3 + score ---------------------
// A = support descriptors (M = m rows), B = query descriptors (N = l cols).
// D fragment gives each lane 8 m-values of ONE query column -> running top-3
// lives entirely in registers (branchless sorted insert); half-waves merged
// once at the end with a shuffle.  No LDS, no branches in the hot loop.
__global__ void sim_topk_k(const f16* __restrict__ qd, const f16* __restrict__ sd,
                           float* __restrict__ scores, int B, int NQ, int Way,
                           int HW, int M, int Mpad) {
  int lane = threadIdx.x & 31;
  int LT = (HW + 15) / 16;                  // 28 query row-tiles
  long gw = ((long)blockIdx.x * blockDim.x + threadIdx.x) >> 5;
  long totalw = (long)B * NQ * Way * LT;
  if (gw >= totalw) return;                 // wave-uniform
  int lt = (int)(gw % LT); long t = gw / LT;
  int w = (int)(t % Way); t /= Way;
  int q = (int)(t % NQ); int b = (int)(t / NQ);
  int col = lane & 15;                      // query column of this lane
  int rsel = lane >> 4;                     // which half of rows / K
  int l = lt * 16 + col;
  int lc = l < HW ? l : HW - 1;
  // B fragment (fixed across the m loop): lane = query col, 16 contiguous K
  const f16* qrow = qd + ((long)(b * NQ + q) * HW + lc) * 64;
  v16h bq0 = *(const v16h*)(qrow + rsel * 16);
  v16h bq1 = *(const v16h*)(qrow + 32 + rsel * 16);

  const f16* sbase = sd + (long)(b * Way + w) * Mpad * 64;
  float t0 = -3.0e38f, t1 = -3.0e38f, t2 = -3.0e38f;
  int MT = Mpad / 16;                       // 138 support tiles
  int mrow = lane & 15;                     // A-fragment row of this lane

  auto step = [&](int mt, bool masked) {
    const f16* srow = sbase + ((long)mt * 16 + mrow) * 64;
    v8h lo0 = *(const v8h*)(srow + rsel * 16);
    v8h hi0 = *(const v8h*)(srow + rsel * 16 + 8);
    v8h lo1 = *(const v8h*)(srow + 32 + rsel * 16);
    v8h hi1 = *(const v8h*)(srow + 40 + rsel * 16);
    v16h a0, a1;
#pragma unroll
    for (int h = 0; h < 8; ++h) {
      a0[h] = lo0[h]; a0[h + 8] = hi0[h];
      a1[h] = lo1[h]; a1[h + 8] = hi1[h];
    }
    v8f c = {};
    c = wmma16x16x32(a0, bq0, c);
    c = wmma16x16x32(a1, bq1, c);
#pragma unroll
    for (int r = 0; r < 8; ++r) {           // lane's 8 m-rows for query `col`
      float v = c[r];
      if (masked && (mt * 16 + rsel * 8 + r) >= M) v = -3.0e38f;
      float d0 = fminf(v, t0); t0 = fmaxf(v, t0);
      float d1 = fminf(d0, t1); t1 = fmaxf(d0, t1);
      t2 = fmaxf(d1, t2);
    }
  };
  for (int mt = 0; mt < MT - 1; ++mt) step(mt, false);   // unmasked fast path
  step(MT - 1, true);                                    // masked tail tile

  // merge the two half-wave top-3 sets (lane ^ 16 owns the other 8 m-rows)
  float u0 = __shfl_xor(t0, 16, 32);
  float u1 = __shfl_xor(t1, 16, 32);
  float u2 = __shfl_xor(t2, 16, 32);
  float vv[3] = {u0, u1, u2};
#pragma unroll
  for (int j = 0; j < 3; ++j) {
    float v = vv[j];
    float d0 = fminf(v, t0); t0 = fmaxf(v, t0);
    float d1 = fminf(d0, t1); t1 = fmaxf(d0, t1);
    t2 = fmaxf(d1, t2);
  }
  if (lane < 16 && l < HW)
    atomicAdd(&scores[(b * NQ + q) * Way + w], t0 + t1 + t2);
}

// ------------------------------- host side ----------------------------------
extern "C" void kernel_launch(void* const* d_in, const int* in_sizes, int n_in,
                              void* d_out, int out_size, void* d_ws, size_t ws_size,
                              hipStream_t stream) {
  (void)in_sizes; (void)n_in; (void)out_size; (void)ws_size;
  const float* query   = (const float*)d_in[0];
  const float* support = (const float*)d_in[1];
  const float* Wt[4] = {(const float*)d_in[2], (const float*)d_in[3],
                        (const float*)d_in[4], (const float*)d_in[5]};
  const float* gs[4] = {(const float*)d_in[6], (const float*)d_in[8],
                        (const float*)d_in[10], (const float*)d_in[12]};
  const float* bs[4] = {(const float*)d_in[7], (const float*)d_in[9],
                        (const float*)d_in[11], (const float*)d_in[13]};

  const int Bb = 2, NQ = 15, Way = 5, Shot = 5, H0 = 84, W0 = 84;
  const int NQall = Bb * NQ;           // 30
  const int NSall = Bb * Way * Shot;   // 50
  const int HWf = 21 * 21;             // 441
  const int M = Shot * HWf;            // 2205
  const int Mpad = 2208;               // padded to 16

  size_t off = 0;
  auto carve = [&](size_t bytes) -> void* {
    off = (off + 255) & ~(size_t)255;
    void* p = (char*)d_ws + off;
    off += bytes;
    return p;
  };
  f16* wp[4];
  wp[0] = (f16*)carve((size_t)4 * 1 * 512 * 2);
  for (int i = 1; i < 4; ++i) wp[i] = (f16*)carve((size_t)4 * 18 * 512 * 2);
  float* stats = (float*)carve(128 * 4);
  f16* in0  = (f16*)carve((size_t)NSall * H0 * W0 * 3 * 2);
  f16* actA = (f16*)carve((size_t)NSall * H0 * W0 * 64 * 2);
  f16* actB = (f16*)carve((size_t)NSall * H0 * W0 * 64 * 2);
  float* cvt = (float*)carve((size_t)NSall * H0 * W0 * 64 * 4);
  f16* qdesc = (f16*)carve((size_t)NQall * HWf * 64 * 2);
  f16* sdesc = (f16*)carve((size_t)Bb * Way * Mpad * 64 * 2);

  // pack weights once per call (deterministic)
  pack_w_k<<<(4 * 1 * 512 + 255) / 256, 256, 0, stream>>>(Wt[0], wp[0], 3);
  for (int i = 1; i < 4; ++i)
    pack_w_k<<<(4 * 18 * 512 + 255) / 256, 256, 0, stream>>>(Wt[i], wp[i], 64);

  auto run_bn = [&](const float* cv, const float* g, const float* b, f16* dst,
                    int N, int H, int W, bool pool) {
    long rows = (long)N * H * W;
    zero_f32_k<<<1, 128, 0, stream>>>(stats, 128);
    long sg = (rows + 3) / 4; if (sg > 1024) sg = 1024;
    bn_stats_k<<<(int)sg, 256, 0, stream>>>(cv, stats, rows);
    long tot = (long)N * (pool ? H / 2 : H) * (pool ? W / 2 : W) * 64;
    int grid = (int)((tot + 255) / 256);
    float ir = 1.f / (float)rows;
    if (pool) bn_act_k<true ><<<grid, 256, 0, stream>>>(cv, stats, g, b, dst, N, H, W, ir);
    else      bn_act_k<false><<<grid, 256, 0, stream>>>(cv, stats, g, b, dst, N, H, W, ir);
  };
  auto conv_grid = [&](int N, int H, int W) {
    long waves = ((long)N * H * W + 15) / 16;
    return (int)((waves + 7) / 8);           // 8 waves / 256-thread block
  };
  auto encode = [&](const float* in_nchw, int N, int isSupport) {
    long tin = (long)N * 3 * H0 * W0;
    to_nhwc_f16_k<<<(int)((tin + 255) / 256), 256, 0, stream>>>(in_nchw, in0, N, 3, H0, W0);
    conv_wmma_k<3><<<conv_grid(N, 84, 84), 256, 0, stream>>>(in0, wp[0], cvt, N, 84, 84);
    run_bn(cvt, gs[0], bs[0], actA, N, 84, 84, true);    // -> N x 42x42x64
    conv_wmma_k<64><<<conv_grid(N, 42, 42), 256, 0, stream>>>(actA, wp[1], cvt, N, 42, 42);
    run_bn(cvt, gs[1], bs[1], actB, N, 42, 42, true);    // -> N x 21x21x64
    conv_wmma_k<64><<<conv_grid(N, 21, 21), 256, 0, stream>>>(actB, wp[2], cvt, N, 21, 21);
    run_bn(cvt, gs[2], bs[2], actA, N, 21, 21, false);
    conv_wmma_k<64><<<conv_grid(N, 21, 21), 256, 0, stream>>>(actA, wp[3], cvt, N, 21, 21);
    run_bn(cvt, gs[3], bs[3], actB, N, 21, 21, false);
    long rows = (long)N * HWf;
    make_desc_k<<<(int)((rows + 255) / 256), 256, 0, stream>>>(
        actB, qdesc, sdesc, N, HWf, isSupport, Way, Shot, Mpad);
  };

  encode(query, NQall, 0);
  long sdn = (long)Bb * Way * Mpad * 64;
  zero_f16_k<<<(int)((sdn + 255) / 256), 256, 0, stream>>>(sdesc, sdn);  // pad cols = 0
  encode(support, NSall, 1);

  zero_f32_k<<<1, 256, 0, stream>>>((float*)d_out, (long)NQall * Way);
  int LT = (HWf + 15) / 16;                        // 28
  long totw = (long)Bb * NQ * Way * LT;            // 4200 waves
  sim_topk_k<<<(int)((totw + 31) / 32), 256, 0, stream>>>(
      qdesc, sdesc, (float*)d_out, Bb, NQ, Way, HWf, M, Mpad);
}
